// PointDiffusionTransformer_26852135534856
// MI455X (gfx1250) — matrix operand order repaired and checked by reading, not compiled, not run on hardware
//
#include <hip/hip_runtime.h>
#include <hip/hip_bf16.h>

// ---------------------------------------------------------------------------
// PointDiffusionTransformer forward on gfx1250 (MI455X, wave32, WMMA).
// Heavy GEMMs: v_wmma_f32_16x16x32_f16 (f16 in, f32 acc).
// A-tile staging: Tensor Data Mover (tensor_load_to_lds + s_wait_tensorcnt)
// when the builtin is available; cooperative loads otherwise.
// ---------------------------------------------------------------------------

typedef __attribute__((ext_vector_type(16))) _Float16 v16h;
typedef __attribute__((ext_vector_type(8)))  _Float16 h8;
typedef __attribute__((ext_vector_type(8)))  float    v8f;
typedef __attribute__((ext_vector_type(4)))  unsigned int u32x4;
typedef __attribute__((ext_vector_type(8)))  int      i32x8;
typedef __attribute__((ext_vector_type(4)))  int      i32x4;

#define BATCH 16
#define NP    4096      // points
#define CIN   302       // channels of x_cat
#define CPAD  320       // 302 padded to multiple of 32
#define CMR   304       // fp32 channel-major row allocation (mult of 16)
#define LRELU_FLAG 1

#if __has_builtin(__builtin_amdgcn_tensor_load_to_lds) && \
    __has_builtin(__builtin_amdgcn_s_wait_tensorcnt)
#define USE_TDM 1
#endif

// ------------------------- WMMA fragment helpers ---------------------------
// A fragment (16x32 f16, row-major source, ISA layout):
//   lanes 0-15 : row = lane,    K = {0..7, 16..23}
//   lanes 16-31: row = lane-16, K = {8..15, 24..31}
__device__ __forceinline__ v16h frag_a(const _Float16* __restrict__ p, int pitch) {
  int lane = threadIdx.x & 31;
  const _Float16* r = p + (long)(lane & 15) * pitch + ((lane >> 4) << 3);
  h8 lo = *(const h8*)(r);
  h8 hi = *(const h8*)(r + 16);
  v16h f;
#pragma unroll
  for (int i = 0; i < 8; ++i) { f[i] = lo[i]; f[i + 8] = hi[i]; }
  return f;
}

// B fragment (32x16 f16) read from a K-TRANSPOSED source: src[col*pitch + k].
//   lanes 0-15 : col = lane,    K = {0..7, 16..23}
//   lanes 16-31: col = lane-16, K = {8..15, 24..31}
__device__ __forceinline__ v16h frag_bT(const _Float16* __restrict__ p, int pitch) {
  int lane = threadIdx.x & 31;
  const _Float16* r = p + (long)(lane & 15) * pitch + ((lane >> 4) << 3);
  h8 lo = *(const h8*)(r);
  h8 hi = *(const h8*)(r + 16);
  v16h f;
#pragma unroll
  for (int i = 0; i < 8; ++i) { f[i] = lo[i]; f[i + 8] = hi[i]; }
  return f;
}

__device__ __forceinline__ v8f wmma16(v16h a, v16h b, v8f c) {
  return __builtin_amdgcn_wmma_f32_16x16x32_f16(false, a, false, b, (short)0, c,
                                                false, false);
}

#ifdef USE_TDM
// Issue a TDM load of a 64x32-half 2D tile (row stride kp halves) into LDS at
// ldsOff, with LDS padding 4 DWORDs every 16 DWORDs (row pitch 32 -> 40 halves).
// Wave-uniform arguments; EXEC is ignored by TENSOR_LOAD_TO_LDS.
__device__ __forceinline__ void tdm_load_a(const _Float16* gp, unsigned ldsOff,
                                           int kp) {
  unsigned long long ga = (unsigned long long)(size_t)gp;
  u32x4 g0;
  g0[0] = 1u;                                   // count=1, user descriptor
  g0[1] = ldsOff;                               // lds_addr (bytes)
  g0[2] = (unsigned)ga;                         // global_addr[31:0]
  g0[3] = (unsigned)((ga >> 32) & 0x1FFFFFFu) | (2u << 30);  // addr[56:32]|type=2
  i32x8 g1;
  g1[0] = (int)((1u << 16)      // data_size = 2 bytes
              | (1u << 20)      // pad_enable
              | (3u << 22)      // pad_interval: 16 DWORDs (one 32-half row)
              | (3u << 25));    // pad_amount: 4 DWORDs (8 halves)
  g1[1] = (int)((unsigned)kp << 16);    // tensor_dim0[15:0] = kp
  g1[2] = (int)(64u << 16);             // tensor_dim0 hi=0 | tensor_dim1 = 64
  g1[3] = (int)(32u << 16);             // tensor_dim1 hi=0 | tile_dim0 = 32
  g1[4] = 64;                           // tile_dim1 = 64 | tile_dim2 = 0
  g1[5] = kp;                           // tensor_dim0_stride[31:0]
  g1[6] = 0;
  g1[7] = 0;
  i32x4 z4 = {0, 0, 0, 0};
#if defined(__clang_major__) && __clang_major__ >= 23
  i32x8 z8 = {0, 0, 0, 0, 0, 0, 0, 0};
  __builtin_amdgcn_tensor_load_to_lds(g0, g1, z4, z4, z8, 0);
#else
  __builtin_amdgcn_tensor_load_to_lds(g0, g1, z4, z4, 0);
#endif
}
#endif

// ------------------------------ build x_cat --------------------------------
__global__ __launch_bounds__(256)
void k_build(const float* __restrict__ x, const float* __restrict__ beta,
             const float* __restrict__ ctx, const int* __restrict__ label,
             float* __restrict__ xc32, _Float16* __restrict__ xc16) {
  long i = (long)blockIdx.x * 256 + threadIdx.x;   // over B*N
  int b = (int)(i / NP), n = (int)(i % NP);
  float bv = beta[b];
  int lb = label[b];
  float sb = sinf(bv), cb = cosf(bv);
#pragma unroll 4
  for (int c = 0; c < CPAD; ++c) {
    float v;
    if      (c < 3)   v = x[i * 3 + c];
    else if (c < 43)  v = (c - 3 == lb) ? 1.f : 0.f;
    else if (c == 43) v = bv;
    else if (c == 44) v = sb;
    else if (c == 45) v = cb;
    else if (c < CIN) v = ctx[b * 256 + (c - 46)];
    else              v = 0.f;
    if (c < CIN) xc32[i * CIN + c] = v;
    xc16[((long)b * CPAD + c) * NP + n] = (_Float16)v;
  }
}

// --------------------------- weight/bias packing ---------------------------
__global__ __launch_bounds__(256)
void k_pack(const float* __restrict__ src, _Float16* __restrict__ dst,
            int M, int K, int Kp, int rowOff, long total) {
  long i = (long)blockIdx.x * 256 + threadIdx.x;
  if (i >= total) return;
  int m = (int)(i / Kp), k = (int)(i % Kp);
  _Float16 v = (m < M && k < K) ? (_Float16)src[(long)m * K + k] : (_Float16)0.f;
  dst[(long)(rowOff + m) * Kp + k] = v;
}

__global__ __launch_bounds__(256)
void k_copy(const float* __restrict__ s, float* __restrict__ d, int n) {
  int i = blockIdx.x * 256 + threadIdx.x;
  if (i < n) d[i] = s[i];
}

// ------------------------------ generic GEMM -------------------------------
// Y[b, outRowOff+m, n] = act( sum_k W[b?][m,k] * X[b, xRowOff+k, n] + bias[m] )
// W: f16 [Mp x Kp] row-major (per-batch stride wStride elements, 0 = shared)
// X: f16 [B, xRows, NP] channel-major.  Block tile 64x128, K-step 32, 8 waves.
__global__ __launch_bounds__(256)
void k_gemm(const _Float16* __restrict__ W, long wStride,
            const _Float16* __restrict__ X, int xRows, int xRowOff,
            const float* __restrict__ bias, int Mvalid,
            float* __restrict__ Y32, int y32Rows,
            _Float16* __restrict__ Y16, int y16Rows,
            int outRowOff, int Kp, int flags) {
  __shared__ _Float16 sA[64 * 40];
  __shared__ _Float16 sB[128 * 40];
  const int b  = blockIdx.z;
  const int m0 = blockIdx.y * 64;
  const int n0 = blockIdx.x * 128;
  const _Float16* Wb = W + (long)b * wStride;
  const _Float16* Xb = X + ((long)b * xRows + xRowOff) * NP;
  const int tid = threadIdx.x;
  const int lane = tid & 31, wave = tid >> 5;
  const int wm = (wave & 1) * 32, wn = (wave >> 1) * 32;

  v8f acc[2][2] = {};
  for (int k0 = 0; k0 < Kp; k0 += 32) {
    // ---- stage A: 64 rows x 32 halves, row pitch 40 ----
#ifdef USE_TDM
    if (wave == 0) {
      // Tensor Data Mover copies the tile; TENSORcnt tracks completion.
      tdm_load_a(Wb + (long)m0 * Kp + k0, (unsigned)(size_t)(&sA[0]), Kp);
      __builtin_amdgcn_s_wait_tensorcnt(0);
    }
#else
    {
      int row = tid >> 2, ch = (tid & 3) * 8;
      h8 v = *(const h8*)(Wb + (long)(m0 + row) * Kp + k0 + ch);
      *(h8*)(sA + row * 40 + ch) = v;
    }
#endif
    // ---- stage B transposed: logical [k][n] -> sB[n*40 + k] ----
#pragma unroll
    for (int r = 0; r < 2; ++r) {
      int kr = (tid >> 4) + r * 16;
      int nc = (tid & 15) * 8;
      h8 v = *(const h8*)(Xb + (long)(k0 + kr) * NP + n0 + nc);
#pragma unroll
      for (int j = 0; j < 8; ++j) sB[(nc + j) * 40 + kr] = v[j];
    }
    if (k0 + 32 < Kp) {   // prefetch next K tile -> global_prefetch_b8
      __builtin_prefetch(Wb + (long)(m0 + (tid >> 2)) * Kp + k0 + 32, 0, 0);
      __builtin_prefetch(Xb + (long)(k0 + 32 + (tid >> 4)) * NP + n0, 0, 0);
    }
    __syncthreads();
#pragma unroll
    for (int mi = 0; mi < 2; ++mi) {
      v16h af = frag_a(sA + (wm + mi * 16) * 40, 40);
#pragma unroll
      for (int ni = 0; ni < 2; ++ni) {
        v16h bf = frag_bT(sB + (wn + ni * 16) * 40, 40);
        acc[mi][ni] = wmma16(af, bf, acc[mi][ni]);
      }
    }
    __syncthreads();
  }
  // epilogue: lane layout of 16x16 f32 tile
  const int nl = lane & 15, mb = (lane >> 4) * 8;
#pragma unroll
  for (int mi = 0; mi < 2; ++mi)
#pragma unroll
    for (int ni = 0; ni < 2; ++ni)
#pragma unroll
      for (int i = 0; i < 8; ++i) {
        int m = m0 + wm + mi * 16 + mb + i;
        int n = n0 + wn + ni * 16 + nl;
        float v = acc[mi][ni][i];
        if (bias && m < Mvalid) v += bias[m];
        if (flags & LRELU_FLAG) v = (v > 0.f) ? v : 0.01f * v;
        if (Y32 && m < Mvalid)
          Y32[((long)b * y32Rows + outRowOff + m) * NP + n] = v;
        if (Y16 && (outRowOff + m) < y16Rows)
          Y16[((long)b * y16Rows + outRowOff + m) * NP + n] = (_Float16)v;
      }
}

// ----------------------- channel attention: scores -------------------------
// scores[b, d1, d2] = scale * sum_n q[b, rowOff+d1, n] * q[b, rowOff+d2, n]
__global__ __launch_bounds__(256)
void k_scores(const _Float16* __restrict__ q, float* __restrict__ sout,
              long sStride, int sOff, int rowOff, int dh, float scale) {
  const int b = blockIdx.y;
  const int wave = threadIdx.x >> 5;
  const int tile = blockIdx.x * 8 + wave;
  const int tpr = dh >> 4;
  if (tile >= tpr * tpr) return;           // wave-uniform
  const int tm = tile / tpr, tn = tile % tpr;
  const _Float16* qa = q + ((long)b * 448 + rowOff + tm * 16) * NP;
  const _Float16* qb = q + ((long)b * 448 + rowOff + tn * 16) * NP;
  v8f acc = {};
  for (int k0 = 0; k0 < NP; k0 += 32) {
    if (k0 + 32 < NP) __builtin_prefetch(qa + k0 + 64, 0, 0);
    v16h af = frag_a(qa + k0, NP);
    v16h bf = frag_bT(qb + k0, NP);     // q rows act as B columns (QK^T, Q==K)
    acc = wmma16(af, bf, acc);
  }
  const int lane = threadIdx.x & 31;
  const int nl = lane & 15, mb = (lane >> 4) * 8;
  float* out = sout + (long)b * sStride + sOff;
#pragma unroll
  for (int i = 0; i < 8; ++i)
    out[(long)(tm * 16 + mb + i) * dh + tn * 16 + nl] = acc[i] * scale;
}

// -------------------------------- softmax ----------------------------------
__global__ __launch_bounds__(256)
void k_softmax(const float* __restrict__ s, _Float16* __restrict__ a,
               long bStride, int off, int dh) {
  __shared__ float red[8];
  const int row = blockIdx.x, b = blockIdx.y, t = threadIdx.x;
  const float* sr = s + (long)b * bStride + off + (long)row * dh;
  float v = (t < dh) ? sr[t] : -3.4e38f;
  float m = v;
  for (int o = 16; o; o >>= 1) m = fmaxf(m, __shfl_xor(m, o));
  if ((t & 31) == 0) red[t >> 5] = m;
  __syncthreads();
  if (t < 8) {
    float mm = red[t];
    for (int o = 4; o; o >>= 1) mm = fmaxf(mm, __shfl_xor(mm, o));
    if (t == 0) red[0] = mm;
  }
  __syncthreads();
  const float mx = red[0];
  __syncthreads();
  float e = (t < dh) ? __expf(v - mx) : 0.f;
  float sum = e;
  for (int o = 16; o; o >>= 1) sum += __shfl_xor(sum, o);
  if ((t & 31) == 0) red[t >> 5] = sum;
  __syncthreads();
  if (t < 8) {
    float ss = red[t];
    for (int o = 4; o; o >>= 1) ss += __shfl_xor(ss, o);
    if (t == 0) red[0] = ss;
  }
  __syncthreads();
  if (t < dh)
    a[(long)b * bStride + off + (long)row * dh + t] = (_Float16)(e / red[0]);
}

// --------------------- fused residual + LayerNorm --------------------------
// v = (a_cm? a_cm : 0) + (b_cm? b_cm : 0) + (a_rm? a_rm : 0); LN over C=302.
__global__ __launch_bounds__(320)
void k_ln(const float* __restrict__ a_cm, const float* __restrict__ b_cm,
          const float* __restrict__ a_rm, const float* __restrict__ g,
          const float* __restrict__ bt, _Float16* __restrict__ out16,
          float* __restrict__ out32) {
  __shared__ float sS[10], sQ[10];
  const int n = blockIdx.x, b = blockIdx.y, c = threadIdx.x;
  float v = 0.f;
  if (c < CIN) {
    if (a_cm) v += a_cm[((long)b * CMR + c) * NP + n];
    if (b_cm) v += b_cm[((long)b * CMR + c) * NP + n];
    if (a_rm) v += a_rm[((long)b * NP + n) * CIN + c];
  }
  float s1 = v, s2 = v * v;
  for (int o = 16; o; o >>= 1) { s1 += __shfl_xor(s1, o); s2 += __shfl_xor(s2, o); }
  if ((c & 31) == 0) { sS[c >> 5] = s1; sQ[c >> 5] = s2; }
  __syncthreads();
  if (c == 0) {
    float a = 0.f, q2 = 0.f;
    for (int i = 0; i < 10; ++i) { a += sS[i]; q2 += sQ[i]; }
    sS[0] = a; sQ[0] = q2;
  }
  __syncthreads();
  const float mean = sS[0] * (1.f / CIN);
  const float var  = sQ[0] * (1.f / CIN) - mean * mean;
  const float r = rsqrtf(var + 1e-5f);
  float y = (c < CIN) ? (v - mean) * r * g[c] + bt[c] : 0.f;
  out16[((long)b * CPAD + c) * NP + n] = (_Float16)y;
  if (out32 && c < CIN) out32[((long)b * CMR + c) * NP + n] = y;
}

// ------------------------------ final output -------------------------------
__global__ __launch_bounds__(256)
void k_out(const float* __restrict__ cm, float* __restrict__ out) {
  long i = (long)blockIdx.x * 256 + threadIdx.x;   // over B*N
  int b = (int)(i / NP), n = (int)(i % NP);
#pragma unroll
  for (int m = 0; m < 3; ++m)
    out[i * 3 + m] = cm[((long)b * CMR + m) * NP + n];
}

// ===========================================================================
//                               host launcher
// ===========================================================================
// Input order (setup_inputs pytree, insertion order, lists flattened):
enum {
  IN_X = 0, IN_BETA, IN_CTX, IN_LABEL,
  IN_EWQK0, IN_EWQK1, IN_EWQK2, IN_EWV0, IN_EWV1, IN_EWV2,
  IN_EBV0, IN_EBV1, IN_EBV2,
  IN_DWQK0, IN_DWQK1, IN_DWQK2, IN_DWV0, IN_DWV1, IN_DWV2,
  IN_DBV0, IN_DBV1, IN_DBV2,
  IN_RH1W, IN_RH1B, IN_LN1G, IN_LN1B,
  IN_CV1W, IN_CV1B, IN_CV2W, IN_CV2B,
  IN_RH2W, IN_RH2B, IN_LN2G, IN_LN2B,
  IN_CV3W, IN_CV3B, IN_CV4W, IN_CV4B,
  IN_LN3G, IN_LN3B, IN_CV5W, IN_CV5B, IN_CV6W, IN_CV6B
};

static const int HD[3]   = {64, 128, 256};
static const int HOFF[3] = {0, 64, 192};
static const int SOFF[3] = {0, 4096, 20480};
#define SSTRIDE 86016L   // per-batch scores/attn elements (64^2+128^2+256^2)

extern "C" void kernel_launch(void* const* d_in, const int* in_sizes, int n_in,
                              void* d_out, int out_size, void* d_ws, size_t ws_size,
                              hipStream_t stream) {
  char* base = (char*)d_ws;
  size_t cur = 0;
  auto alloc = [&](size_t bytes) -> char* {
    char* p = base + cur;
    cur = (cur + bytes + 255) & ~(size_t)255;
    return p;
  };
  const long BN = (long)BATCH * NP;

  float*     XC32 = (float*)    alloc(BN * CIN * 4);            // x_cat row-major
  _Float16*  XC16 = (_Float16*) alloc(BN * CPAD * 2);           // x_cat chan-major f16
  _Float16*  Q16  = (_Float16*) alloc(BN * 448 * 2);            // q / attn-out
  _Float16*  V16  = (_Float16*) alloc(BN * 448 * 2);            // v (biased)
  float*     S32  = (float*)    alloc((long)BATCH * SSTRIDE * 4);
  _Float16*  A16  = (_Float16*) alloc((long)BATCH * SSTRIDE * 2);
  float*     CM32 = (float*)    alloc(BN * CMR * 4);            // f32 chan-major scratch
  float*     XD32 = (float*)    alloc(BN * CMR * 4);            // xd residual / final
  _Float16*  H320 = (_Float16*) alloc(BN * CPAD * 2);
  _Float16*  H512 = (_Float16*) alloc(BN * 512 * 2);
  _Float16*  H128 = (_Float16*) alloc(BN * 128 * 2);
  _Float16*  WENCQ = (_Float16*)alloc(448L * CPAD * 2);
  _Float16*  WENCV = (_Float16*)alloc(448L * CPAD * 2);
  _Float16*  WDECQ = (_Float16*)alloc(448L * CPAD * 2);
  _Float16*  WDECV = (_Float16*)alloc(448L * CPAD * 2);
  _Float16*  WRH1  = (_Float16*)alloc(320L * 448 * 2);
  _Float16*  WRH2  = (_Float16*)alloc(320L * 448 * 2);
  _Float16*  WCV1  = (_Float16*)alloc(512L * CPAD * 2);
  _Float16*  WCV2  = (_Float16*)alloc(320L * 512 * 2);
  _Float16*  WCV3  = (_Float16*)alloc(512L * CPAD * 2);
  _Float16*  WCV4  = (_Float16*)alloc(320L * 512 * 2);
  _Float16*  WCV5  = (_Float16*)alloc(128L * CPAD * 2);
  _Float16*  WCV6  = (_Float16*)alloc(64L * 128 * 2);
  float*     BENC  = (float*)   alloc(448 * 4);
  float*     BDEC  = (float*)   alloc(448 * 4);

  auto pack = [&](int idx, _Float16* dst, int M, int K, int Kp, int rowOff, int rows) {
    long total = (long)rows * Kp;
    k_pack<<<dim3((unsigned)((total + 255) / 256)), dim3(256), 0, stream>>>(
        (const float*)d_in[idx], dst, M, K, Kp, rowOff, total);
  };
  auto gemm = [&](const _Float16* W, long wStride, int MpAlloc,
                  const _Float16* X, int xRows, int xRowOff, int Kp,
                  const float* bias, int Mvalid,
                  float* Y32, int y32Rows, _Float16* Y16, int y16Rows,
                  int outRowOff, int flags) {
    dim3 g(NP / 128, MpAlloc / 64, BATCH);
    k_gemm<<<g, dim3(256), 0, stream>>>(W, wStride, X, xRows, xRowOff, bias,
                                        Mvalid, Y32, y32Rows, Y16, y16Rows,
                                        outRowOff, Kp, flags);
  };

  // ---- pack weights (f32 -> zero-padded f16) ----
  for (int h = 0; h < 3; ++h) {
    pack(IN_EWQK0 + h, WENCQ, HD[h], CIN, CPAD, HOFF[h], HD[h]);
    pack(IN_EWV0  + h, WENCV, HD[h], CIN, CPAD, HOFF[h], HD[h]);
    pack(IN_DWQK0 + h, WDECQ, HD[h], CIN, CPAD, HOFF[h], HD[h]);
    pack(IN_DWV0  + h, WDECV, HD[h], CIN, CPAD, HOFF[h], HD[h]);
    k_copy<<<dim3((HD[h] + 255) / 256), dim3(256), 0, stream>>>(
        (const float*)d_in[IN_EBV0 + h], BENC + HOFF[h], HD[h]);
    k_copy<<<dim3((HD[h] + 255) / 256), dim3(256), 0, stream>>>(
        (const float*)d_in[IN_DBV0 + h], BDEC + HOFF[h], HD[h]);
  }
  pack(IN_RH1W, WRH1, CIN, 448, 448, 0, 320);
  pack(IN_RH2W, WRH2, CIN, 448, 448, 0, 320);
  pack(IN_CV1W, WCV1, 512, CIN, CPAD, 0, 512);
  pack(IN_CV2W, WCV2, CIN, 512, 512, 0, 320);
  pack(IN_CV3W, WCV3, 512, CIN, CPAD, 0, 512);
  pack(IN_CV4W, WCV4, CIN, 512, 512, 0, 320);
  pack(IN_CV5W, WCV5, 128, CIN, CPAD, 0, 128);
  pack(IN_CV6W, WCV6, 3, 128, 128, 0, 64);

  // ---- x_cat ----
  k_build<<<dim3((unsigned)(BN / 256)), dim3(256), 0, stream>>>(
      (const float*)d_in[IN_X], (const float*)d_in[IN_BETA],
      (const float*)d_in[IN_CTX], (const int*)d_in[IN_LABEL], XC32, XC16);

  auto attention = [&](const _Float16* WQ, const _Float16* WV, const float* bv,
                       const _Float16* Xq, int xqRows) {
    // q = Wq @ Xq ; v = Wv @ x_cat + bv       (both f16 chan-major [B,448,N])
    gemm(WQ, 0, 448, Xq, xqRows, 0, CPAD, nullptr, 448, nullptr, 0, Q16, 448, 0, 0);
    gemm(WV, 0, 448, XC16, CPAD, 0, CPAD, bv, 448, nullptr, 0, V16, 448, 0, 0);
    for (int h = 0; h < 3; ++h) {
      int tiles = (HD[h] / 16) * (HD[h] / 16);
      k_scores<<<dim3((tiles + 7) / 8, BATCH), dim3(256), 0, stream>>>(
          Q16, S32, SSTRIDE, SOFF[h], HOFF[h], HD[h],
          rsqrtf((float)HD[h]));
      k_softmax<<<dim3(HD[h], BATCH), dim3(256), 0, stream>>>(
          S32, A16, SSTRIDE, SOFF[h], HD[h]);
      // out_head = A @ V_head   (A per-batch: stride SSTRIDE) -> reuse Q16
      gemm(A16 + SOFF[h], SSTRIDE, HD[h], V16, 448, HOFF[h], HD[h],
           nullptr, HD[h], nullptr, 0, Q16, 448, HOFF[h], 0);
    }
  };

  // ================= encoder =================
  attention(WENCQ, WENCV, BENC, XC16, CPAD);
  // rh1: [302,448] @ attn -> f32 chan-major
  gemm(WRH1, 0, 320, Q16, 448, 0, 448, (const float*)d_in[IN_RH1B], CIN,
       CM32, CMR, nullptr, 0, 0, 0);
  // ln1(x_cat + rh1) -> f16 chan-major
  k_ln<<<dim3(NP, BATCH), dim3(320), 0, stream>>>(
      CM32, nullptr, XC32, (const float*)d_in[IN_LN1G],
      (const float*)d_in[IN_LN1B], H320, nullptr);
  // cv1 (lrelu) -> 512 ; cv2 (lrelu) -> 302 (f16, padded rows are zero)
  gemm(WCV1, 0, 512, H320, CPAD, 0, CPAD, (const float*)d_in[IN_CV1B], 512,
       nullptr, 0, H512, 512, 0, LRELU_FLAG);
  gemm(WCV2, 0, 320, H512, 512, 0, 512, (const float*)d_in[IN_CV2B], CIN,
       nullptr, 0, H320, CPAD, 0, LRELU_FLAG);

  // ================= decoder =================
  attention(WDECQ, WDECV, BDEC, H320, CPAD);
  gemm(WRH2, 0, 320, Q16, 448, 0, 448, (const float*)d_in[IN_RH2B], CIN,
       CM32, CMR, nullptr, 0, 0, 0);
  // ln2(x_cat + rh2) -> xd (f16 for cv3, f32 for residual)
  k_ln<<<dim3(NP, BATCH), dim3(320), 0, stream>>>(
      CM32, nullptr, XC32, (const float*)d_in[IN_LN2G],
      (const float*)d_in[IN_LN2B], H320, XD32);
  gemm(WCV3, 0, 512, H320, CPAD, 0, CPAD, (const float*)d_in[IN_CV3B], 512,
       nullptr, 0, H512, 512, 0, LRELU_FLAG);
  gemm(WCV4, 0, 320, H512, 512, 0, 512, (const float*)d_in[IN_CV4B], CIN,
       CM32, CMR, nullptr, 0, 0, LRELU_FLAG);
  // ln3(xdn + xd) -> f16
  k_ln<<<dim3(NP, BATCH), dim3(320), 0, stream>>>(
      CM32, XD32, nullptr, (const float*)d_in[IN_LN3G],
      (const float*)d_in[IN_LN3B], H320, nullptr);
  // cv5 (lrelu) -> 128 ; cv6 -> 3 (f32 chan-major into XD32)
  gemm(WCV5, 0, 128, H320, CPAD, 0, CPAD, (const float*)d_in[IN_CV5B], 128,
       nullptr, 0, H128, 128, 0, LRELU_FLAG);
  gemm(WCV6, 0, 64, H128, 128, 0, 128, (const float*)d_in[IN_CV6B], 3,
       XD32, CMR, nullptr, 0, 0, 0);
  // transpose [B,3,N] -> [B,N,3]
  k_out<<<dim3((unsigned)(BN / 256)), dim3(256), 0, stream>>>(XD32, (float*)d_out);
}
